// SelfAttention_18700287607690
// MI455X (gfx1250) — matrix operand order repaired and checked by reading, not compiled, not run on hardware
//
#include <hip/hip_runtime.h>
#include <hip/hip_bf16.h>

#define BB  8
#define NN  4096
#define CC  256
#define DQK 32
#define DV  128
#define VPAD 40   // padded LDS row stride (elements) for staged tiles

typedef __attribute__((ext_vector_type(16))) __bf16 v16bf;
typedef __attribute__((ext_vector_type(8)))  __bf16 v8bf;
typedef __attribute__((ext_vector_type(8)))  float  v8f;

static __device__ __forceinline__ unsigned lane_id() { return threadIdx.x & 31u; }

static __device__ __forceinline__ v8f wmma_bf16(v16bf a, v16bf b, v8f c) {
  return __builtin_amdgcn_wmma_f32_16x16x32_bf16(false, a, false, b, (short)0, c,
                                                 false, false);
}

// Async DMA of 16 bytes global -> LDS (gfx1250, tracked by ASYNCcnt).
// LDS address = low 32 bits of the generic pointer.
static __device__ __forceinline__ void async_b128(const void* g, const void* l) {
  unsigned lds = (unsigned)(size_t)l;
  asm volatile("global_load_async_to_lds_b128 %0, %1, off"
               :: "v"(lds), "v"(g) : "memory");
}
static __device__ __forceinline__ void wait_async0() {
  asm volatile("s_wait_asynccnt 0x0" ::: "memory");
}

// A tile (16 rows x K=32) from bf16 row-major storage, row stride rs (elements).
// ISA A-layout: lanes 0-15 hold row m, K={0..7, 16..23}; lanes 16-31 K={8..15, 24..31}.
static __device__ __forceinline__ v16bf load_A_bf16(const __bf16* base, int rs) {
  unsigned l = lane_id(), m = l & 15u, h = l >> 4;
  const __bf16* p = base + (size_t)m * rs + 8u * h;
  v8bf c0 = *(const v8bf*)(p);
  v8bf c1 = *(const v8bf*)(p + 16);
  v16bf a;
#pragma unroll
  for (int i = 0; i < 8; ++i) { a[i] = c0[i]; a[i + 8] = c1[i]; }
  return a;
}

// A tile from f32 row-major storage, converted to bf16.
static __device__ __forceinline__ v16bf load_A_f32(const float* base, int rs) {
  unsigned l = lane_id(), m = l & 15u, h = l >> 4;
  const float* p = base + (size_t)m * rs + 8u * h;
  v16bf a;
#pragma unroll
  for (int i = 0; i < 8; ++i) { a[i] = (__bf16)p[i]; a[i + 8] = (__bf16)p[i + 16]; }
  return a;
}

// B tile (K=32 x 16 cols) loaded from B^T stored row-major [16][>=32], stride rs.
// B-layout: lane n<16 -> column n, K={0..7, 16..23} ; lane n+16 -> column n, K={8..15, 24..31}.
static __device__ __forceinline__ v16bf load_B_bf16(const __bf16* baseT, int rs) {
  unsigned l = lane_id(), n = l & 15u, h = l >> 4;
  const __bf16* p = baseT + (size_t)n * rs + 8u * h;
  v8bf c0 = *(const v8bf*)(p);
  v8bf c1 = *(const v8bf*)(p + 16);
  v16bf b;
#pragma unroll
  for (int i = 0; i < 8; ++i) { b[i] = c0[i]; b[i + 8] = c1[i]; }
  return b;
}

// ---------------- Kernel 1: weight transpose + bf16 convert ----------------
__global__ void prep_weights(const float* Wf, const float* Wg, const float* Wh,
                             const float* Wo, __bf16* WfT, __bf16* WgT,
                             __bf16* WhT, __bf16* WoT) {
  int i = blockIdx.x * blockDim.x + threadIdx.x;
  if (i < 8192) {                       // WfT[32][256] = Wf[256][32]^T
    int d = i >> 8, c = i & 255;
    WfT[i] = (__bf16)Wf[c * DQK + d];
  } else if (i < 16384) {
    int j = i - 8192; int d = j >> 8, c = j & 255;
    WgT[j] = (__bf16)Wg[c * DQK + d];
  } else if (i < 49152) {               // WhT[128][256]
    int j = i - 16384; int d = j >> 8, c = j & 255;
    WhT[j] = (__bf16)Wh[c * DV + d];
  } else if (i < 81920) {               // WoT[256][128] = Wo[128][256]^T
    int j = i - 49152; int c = j >> 7, d = j & 127;
    WoT[j] = (__bf16)Wo[d * CC + c];
  }
}

// ---------------- Kernel 2: fused QKV projections ----------------
// fbuf,gbuf: [B][N][32] bf16 row-major ; hT: [B][128][N] bf16 (V transposed)
__global__ __launch_bounds__(256) void proj_kernel(
    const float* __restrict__ x, const __bf16* __restrict__ WfT,
    const __bf16* __restrict__ WgT, const __bf16* __restrict__ WhT,
    const float* __restrict__ bf_, const float* __restrict__ bg_,
    const float* __restrict__ bh_, __bf16* __restrict__ fbuf,
    __bf16* __restrict__ gbuf, __bf16* __restrict__ hT) {
  __shared__ __bf16 ldsT[8][16][16];
  int w = threadIdx.x >> 5;
  int unit = blockIdx.x * 8 + w;            // 2048 pixel-tiles * 12 out-tiles
  int ot = unit % 12;
  int ptl = unit / 12;
  int b = ptl >> 8, qt = ptl & 255;
  unsigned l = lane_id(), n = l & 15u, h = l >> 4;

  const __bf16* WT; const float* bias; int ct;
  if (ot < 2)      { WT = WfT; bias = bf_; ct = ot; }
  else if (ot < 4) { WT = WgT; bias = bg_; ct = ot - 2; }
  else             { WT = WhT; bias = bh_; ct = ot - 4; }

  const float* xbase = x + ((size_t)b * NN + (size_t)qt * 16) * CC;
  v8f acc = {};
#pragma unroll
  for (int kk = 0; kk < 8; ++kk) {
    v16bf A  = load_A_f32(xbase + kk * 32, CC);
    v16bf Bm = load_B_bf16(WT + (size_t)(ct * 16) * CC + kk * 32, CC);
    acc = wmma_bf16(A, Bm, acc);
  }
  float bv = bias[ct * 16 + n];
#pragma unroll
  for (int r = 0; r < 8; ++r) acc[r] += bv;

  if (ot < 4) {
    // C-layout -> row-major via per-wave LDS transpose, coalesced b128 store
#pragma unroll
    for (int r = 0; r < 8; ++r) ldsT[w][r + 8 * h][n] = (__bf16)acc[r];
    asm volatile("" ::: "memory");        // LDS is in-order within a wave
    __bf16* out = (ot < 2) ? fbuf : gbuf;
    int row = l >> 1, c0 = (l & 1) * 8;
    v8bf vv = *(const v8bf*)&ldsT[w][row][c0];
    *(v8bf*)(out + ((size_t)b * NN + (size_t)qt * 16 + row) * DQK + ct * 16 + c0) = vv;
  } else {
    // transposed store: lane holds dim n, 8 consecutive pixels -> one b128
    v8bf vv;
#pragma unroll
    for (int r = 0; r < 8; ++r) vv[r] = (__bf16)acc[r];
    *(v8bf*)(hT + ((size_t)b * DV + ct * 16 + n) * NN + (size_t)qt * 16 + 8 * h) = vv;
  }
}

// ---------------- Kernel 3: flash attention with async-LDS streaming ----------------
// Block = 8 waves = 8 query tiles of one batch. Per 32-key step, the block stages
// f (32x32) and V^T (128x32) into LDS once via GLOBAL_LOAD_ASYNC_TO_LDS_B128
// (double buffered; async engine fills tile k+1 while WMMAs consume tile k).
__global__ __launch_bounds__(256) void attn_kernel(
    const __bf16* __restrict__ fbuf, const __bf16* __restrict__ gbuf,
    const __bf16* __restrict__ hT, __bf16* __restrict__ obuf) {
  __shared__ __align__(16) __bf16 ldsP[8][16][32];   // per-wave P tiles
  // streaming buffers overlap the (post-loop) output staging region
  __shared__ __align__(16) union SU {
    struct { __bf16 F[2][32][VPAD]; __bf16 V[2][128][VPAD]; } s;  // 25.6 KB
    __bf16 O[8][16][128];                                         // 32 KB
  } u;

  int tid = threadIdx.x;
  int w = tid >> 5;
  int unit = blockIdx.x * 8 + w;        // 8 query-tiles of same batch per block
  int b = unit >> 8, qt = unit & 255;
  unsigned l = lane_id(), n = l & 15u, h = l >> 4;

  const __bf16* fB = fbuf + (size_t)b * NN * DQK;
  const __bf16* hB = hT + (size_t)b * DV * NN;

  v16bf Ag = load_A_bf16(gbuf + ((size_t)b * NN + (size_t)qt * 16) * DQK, DQK);

  v8f O[8] = {};
  float mrun[8], lrun[8];
#pragma unroll
  for (int r = 0; r < 8; ++r) { mrun[r] = -1e30f; lrun[r] = 0.0f; }

  // cooperative async stage of one 32-key block into buffer `bn`
  auto issue_stage = [&](int bn, int kb) {
    if (tid < 128) {                    // f tile: 32 rows x 4 x b128
      int row = tid >> 2, ch = tid & 3;
      async_b128(fB + (size_t)(kb + row) * DQK + ch * 8, &u.s.F[bn][row][ch * 8]);
    }
#pragma unroll
    for (int j = 0; j < 2; ++j) {       // V^T tile: 128 rows x 4 x b128
      int idx = j * 256 + tid;
      int row = idx >> 2, ch = idx & 3;
      async_b128(hB + (size_t)row * NN + kb + ch * 8, &u.s.V[bn][row][ch * 8]);
    }
  };

  issue_stage(0, 0);
  for (int it = 0; it < NN / 32; ++it) {
    int bn = it & 1;
    wait_async0();                      // my async fills complete
    __syncthreads();                    // => everyone's fills visible to all waves
    if (it + 1 < NN / 32) issue_stage(bn ^ 1, (it + 1) * 32);  // overlap next stage

    // scores for 32 keys: d_qk=32 is exactly one bf16 WMMA K-step
    v16bf B0 = load_B_bf16(&u.s.F[bn][0][0], VPAD);
    v16bf B1 = load_B_bf16(&u.s.F[bn][16][0], VPAD);
    v8f z0 = {}, z1 = {};
    v8f s0 = wmma_bf16(Ag, B0, z0);
    v8f s1 = wmma_bf16(Ag, B1, z1);

    // online softmax: row stats reduced across the 16 lanes sharing a row group
#pragma unroll
    for (int r = 0; r < 8; ++r) {
      float mx = fmaxf(s0[r], s1[r]);
#pragma unroll
      for (int off = 8; off >= 1; off >>= 1) mx = fmaxf(mx, __shfl_xor(mx, off, 32));
      float mnew = fmaxf(mrun[r], mx);
      float sc = __expf(mrun[r] - mnew);
      float p0 = __expf(s0[r] - mnew);
      float p1 = __expf(s1[r] - mnew);
      float sum = p0 + p1;
#pragma unroll
      for (int off = 8; off >= 1; off >>= 1) sum += __shfl_xor(sum, off, 32);
      lrun[r] = lrun[r] * sc + sum;
      mrun[r] = mnew;
#pragma unroll
      for (int t = 0; t < 8; ++t) O[t][r] *= sc;
      ldsP[w][r + 8 * h][n]      = (__bf16)p0;
      ldsP[w][r + 8 * h][16 + n] = (__bf16)p1;
    }
    asm volatile("" ::: "memory");
    v16bf Ap = load_A_bf16(&ldsP[w][0][0], 32);   // C-layout -> A-layout via LDS
#pragma unroll
    for (int t = 0; t < 8; ++t) {                 // O(16x128) += P(16x32) x V(32x128)
      v16bf Bv = load_B_bf16(&u.s.V[bn][t * 16][0], VPAD);
      O[t] = wmma_bf16(Ap, Bv, O[t]);
    }
    __syncthreads();                    // all waves done with buffer bn
  }

  // normalize, stage to LDS (reuses streaming buffers), coalesced bf16 store
#pragma unroll
  for (int r = 0; r < 8; ++r) {
    float inv = 1.0f / lrun[r];
#pragma unroll
    for (int t = 0; t < 8; ++t)
      u.O[w][r + 8 * h][t * 16 + n] = (__bf16)(O[t][r] * inv);
  }
  asm volatile("" ::: "memory");
  int row = l >> 1, c0 = (l & 1) * 64;
#pragma unroll
  for (int j = 0; j < 8; ++j) {
    v8bf vv = *(const v8bf*)&u.O[w][row][c0 + j * 8];
    *(v8bf*)(obuf + ((size_t)b * NN + (size_t)qt * 16 + row) * DV + c0 + j * 8) = vv;
  }
}

// ---------------- Kernel 4: output projection + gamma*o + residual ----------------
__global__ __launch_bounds__(256) void outproj_kernel(
    const float* __restrict__ x, const __bf16* __restrict__ obuf,
    const __bf16* __restrict__ WoT, const float* __restrict__ bo,
    const float* __restrict__ gptr, float* __restrict__ out) {
  __shared__ float ldsD[8][16][16];
  int w = threadIdx.x >> 5;
  int unit = blockIdx.x * 8 + w;        // 2048 pixel-tiles * 16 outC-tiles
  int ct = unit & 15;
  int ptl = unit >> 4;
  int b = ptl >> 8, qt = ptl & 255;
  unsigned l = lane_id(), n = l & 15u, h = l >> 4;
  float gamma = gptr[0];

  const __bf16* ob = obuf + ((size_t)b * NN + (size_t)qt * 16) * DV;
  v8f acc = {};
#pragma unroll
  for (int kk = 0; kk < 4; ++kk) {      // K loop over d_v=128
    v16bf A  = load_A_bf16(ob + kk * 32, DV);
    v16bf Bm = load_B_bf16(WoT + (size_t)(ct * 16) * DV + kk * 32, DV);
    acc = wmma_bf16(A, Bm, acc);
  }
  float bv = bo[ct * 16 + n];
#pragma unroll
  for (int r = 0; r < 8; ++r) ldsD[w][r + 8 * h][n] = acc[r] + bv;
  asm volatile("" ::: "memory");

  int row = l >> 1, c0 = (l & 1) * 8;
  size_t base = ((size_t)b * NN + (size_t)qt * 16 + row) * CC + ct * 16 + c0;
  const float* xrow = x + base;
  float* orow = out + base;
#pragma unroll
  for (int j = 0; j < 2; ++j) {
    float4 xd = *(const float4*)(xrow + j * 4);
    float4 dd = *(const float4*)&ldsD[w][row][c0 + j * 4];
    float4 rr;
    rr.x = gamma * dd.x + xd.x;
    rr.y = gamma * dd.y + xd.y;
    rr.z = gamma * dd.z + xd.z;
    rr.w = gamma * dd.w + xd.w;
    *(float4*)(orow + j * 4) = rr;
  }
}

extern "C" void kernel_launch(void* const* d_in, const int* in_sizes, int n_in,
                              void* d_out, int out_size, void* d_ws, size_t ws_size,
                              hipStream_t stream) {
  (void)in_sizes; (void)n_in; (void)out_size; (void)ws_size;
  const float* x   = (const float*)d_in[0];
  const float* Wf  = (const float*)d_in[1];
  const float* bf_ = (const float*)d_in[2];
  const float* Wg  = (const float*)d_in[3];
  const float* bg_ = (const float*)d_in[4];
  const float* Wh  = (const float*)d_in[5];
  const float* bh_ = (const float*)d_in[6];
  const float* Wo  = (const float*)d_in[7];
  const float* bo_ = (const float*)d_in[8];
  const float* gm  = (const float*)d_in[9];
  float* out = (float*)d_out;

  char* ws = (char*)d_ws;
  __bf16* WfT  = (__bf16*)(ws);                    //  16 KB
  __bf16* WgT  = (__bf16*)(ws + 16384);            //  16 KB
  __bf16* WhT  = (__bf16*)(ws + 32768);            //  64 KB
  __bf16* WoT  = (__bf16*)(ws + 98304);            //  64 KB
  __bf16* fbuf = (__bf16*)(ws + 163840);           //   2 MB
  __bf16* gbuf = (__bf16*)(ws + 163840 + 2097152); //   2 MB
  __bf16* hT   = (__bf16*)(ws + 163840 + 2 * 2097152);            // 8 MB
  __bf16* obuf = (__bf16*)(ws + 163840 + 2 * 2097152 + 8388608);  // 8 MB

  prep_weights<<<320, 256, 0, stream>>>(Wf, Wg, Wh, Wo, WfT, WgT, WhT, WoT);
  proj_kernel<<<3072, 256, 0, stream>>>(x, WfT, WgT, WhT, bf_, bg_, bh_,
                                        fbuf, gbuf, hT);
  attn_kernel<<<256, 256, 0, stream>>>(fbuf, gbuf, hT, obuf);
  outproj_kernel<<<4096, 256, 0, stream>>>(x, obuf, WoT, bo_, gm, out);
}
